// DS2_996432412756
// MI455X (gfx1250) — matrix operand order repaired and checked by reading, not compiled, not run on hardware
//
#include <hip/hip_runtime.h>

// ---------------------------------------------------------------------------
// Types for CDNA5 WMMA (wave32, v_wmma_f32_16x16x32_bf16) and TDM
// ---------------------------------------------------------------------------
typedef __attribute__((ext_vector_type(16))) __bf16         v16bf;
typedef __attribute__((ext_vector_type(16))) unsigned short v16u;
typedef __attribute__((ext_vector_type(8)))  unsigned short v8u;
typedef __attribute__((ext_vector_type(8)))  float          v8f;
typedef __attribute__((ext_vector_type(4)))  unsigned int   u32x4;
typedef __attribute__((ext_vector_type(8)))  int            i32x8;
typedef __attribute__((ext_vector_type(4)))  int            i32x4;

#define BN_EPS 1e-3f

// float -> bf16 (round to nearest even), pure integer.
__device__ __forceinline__ unsigned short f2bf(float f) {
    unsigned u = __builtin_bit_cast(unsigned, f);
    unsigned r = u + 0x7FFFu + ((u >> 16) & 1u);
    return (unsigned short)(r >> 16);
}

// Merge two 8-half runs into one 16-half WMMA operand register block.
__device__ __forceinline__ v16bf mk_frag(v8u lo, v8u hi) {
    v16u v = __builtin_shufflevector(lo, hi, 0, 1, 2, 3, 4, 5, 6, 7,
                                     8, 9, 10, 11, 12, 13, 14, 15);
    return __builtin_bit_cast(v16bf, v);
}

// LDS byte offset of a __shared__ object: per ISA, generic LDS address has the
// LDS offset in addr[31:0].
__device__ __forceinline__ unsigned lds_byte_offset(const void* p) {
    return (unsigned)(unsigned long long)p;
}

// ---------------------------------------------------------------------------
// TDM: 2D bf16 tile load (rows x cols halves) from global (row stride ldn
// halves) into LDS with 8-half row padding (LDS row stride = cols+8 halves).
// Issue from ONE wave only; follow with s_wait_tensorcnt + barrier.
// ---------------------------------------------------------------------------
#if __has_builtin(__builtin_amdgcn_tensor_load_to_lds)
#define HAVE_TDM 1
__device__ __forceinline__ void tdm_load_tile_pad8(
    const unsigned short* gsrc, unsigned lds_addr, int rows, int cols, int ldn)
{
    unsigned long long ga = (unsigned long long)gsrc;
    u32x4 g0;
    g0[0] = 1u;                                   // count=1 valid descriptor
    g0[1] = lds_addr;                             // lds_addr [63:32]
    g0[2] = (unsigned)(ga & 0xFFFFFFFFu);         // global_addr [95:64]
    g0[3] = (unsigned)((ga >> 32) & 0x01FFFFFFu)  // global_addr [120:96]
          | (2u << 30);                           // type=2 (image)

    // pad: interval code 3 = 16 dwords (one 32-half row), amount code 3 = 4 dw
    i32x8 g1;
    g1[0] = (int)((1u << 16)      // data_size = 2 bytes
                | (1u << 20)      // pad_enable
                | (3u << 22)      // pad_interval
                | (3u << 25));    // pad_amount
    g1[1] = 0;                    // abar=0, tensor_dim0[15:0]=0 (dim0=2^30)
    g1[2] = 0x4000;               // tensor_dim0[31:16]=0x4000, dim1 lo=0
    g1[3] = (int)(0x4000u | ((unsigned)cols << 16));  // dim1 hi, tile_dim0
    g1[4] = rows;                 // tile_dim1 (tile_dim2 = 0)
    g1[5] = ldn;                  // tensor_dim0_stride[31:0]
    g1[6] = 0;
    g1[7] = 0;

    i32x4 z4 = {0, 0, 0, 0};
#if defined(__clang_major__) && __clang_major__ >= 23
    i32x8 z8 = {0, 0, 0, 0, 0, 0, 0, 0};
    __builtin_amdgcn_tensor_load_to_lds(g0, g1, z4, z4, z8, 0);
#else
    __builtin_amdgcn_tensor_load_to_lds(g0, g1, z4, z4, 0);
#endif
}
#else
#define HAVE_TDM 0
#endif

// ---------------------------------------------------------------------------
// Small utility kernels
// ---------------------------------------------------------------------------
__global__ void cvt_f32_bf16_kernel(const float* __restrict__ in,
                                    unsigned short* __restrict__ out, size_t n) {
    size_t i = (size_t)blockIdx.x * blockDim.x + threadIdx.x;
    if (i < n) out[i] = f2bf(in[i]);
}

__global__ void bn_cvt_kernel(const float* __restrict__ in,
                              unsigned short* __restrict__ out,
                              const float* __restrict__ gamma,
                              const float* __restrict__ beta,
                              const float* __restrict__ mean,
                              const float* __restrict__ var,
                              size_t n, int C) {
    size_t i = (size_t)blockIdx.x * blockDim.x + threadIdx.x;
    if (i < n) {
        int c = (int)(i & (size_t)(C - 1));
        float v = gamma[c] * (in[i] - mean[c]) * rsqrtf(var[c] + BN_EPS) + beta[c];
        out[i] = f2bf(v);
    }
}

__global__ void zero_f32_kernel(float* __restrict__ p, size_t n) {
    size_t i = (size_t)blockIdx.x * blockDim.x + threadIdx.x;
    if (i < n) p[i] = 0.0f;
}

__global__ void zero_u16_kernel(unsigned short* __restrict__ p, size_t n) {
    size_t i = (size_t)blockIdx.x * blockDim.x + threadIdx.x;
    if (i < n) p[i] = 0;
}

// ---------------------------------------------------------------------------
// Generic bf16 WMMA GEMM: Out[M,N] = epilogue(A[M,K] @ B[K,N] + bias)
//   mode 0: A bf16 row-major (lda = K); A tile staged by the TDM.
//   mode 1: implicit im2col of conv input x (32,1024,80), SAME pad 4/5, K pad
//           880 -> 896.
// Block 256 threads = 8 waves; tile 32(M) x 64(N); K chunk 32.
// B is staged TRANSPOSED so both fragments load as contiguous ds_load_b128.
// ---------------------------------------------------------------------------
#define EF_BIAS     1
#define EF_RELU     2
#define EF_BN       4
#define EF_OUT_BF16 8

__global__ __launch_bounds__(256) void wmma_gemm_kernel(
    const unsigned short* __restrict__ A,
    const float* __restrict__ Xconv,
    const unsigned short* __restrict__ Bm,
    const float* __restrict__ bias,
    float* __restrict__ Cf,
    unsigned short* __restrict__ Cb,
    int M, int N, int K, int lda, int ldb,
    int mode, int flags,
    const float* __restrict__ gamma, const float* __restrict__ beta,
    const float* __restrict__ mean,  const float* __restrict__ var)
{
    __shared__ unsigned short As[32][32 + 8];     // [m][k], rows 16B-aligned
    __shared__ unsigned short Bt[64][32 + 8];     // [n][k] transposed

    const int tid  = threadIdx.x;
    const int lane = tid & 31;
    const int wave = tid >> 5;       // 0..7
    const int wm   = wave >> 2;      // M sub-tile 0..1
    const int wn   = wave & 3;       // N sub-tile 0..3
    const int m0   = blockIdx.y * 32;
    const int n0   = blockIdx.x * 64;

    const int am  = wm * 16 + (lane & 15);
    const int akb = (lane >> 4) << 3;            // 0 / 8
    const int bn_ = wn * 16 + (lane & 15);
    const int bkb = (lane >> 4) << 4;            // 0 / 16

    v8f acc = {0.f, 0.f, 0.f, 0.f, 0.f, 0.f, 0.f, 0.f};

    const unsigned as_lds = lds_byte_offset(&As[0][0]);
    const int nk = K >> 5;
    for (int kc = 0; kc < nk; ++kc) {
        const int k0 = kc << 5;
        // ---- stage A ----
        if (mode == 0) {
#if HAVE_TDM
            if (wave == 0)
                tdm_load_tile_pad8(&A[(size_t)m0 * lda + k0], as_lds, 32, 32, lda);
#else
            #pragma unroll
            for (int it = 0; it < 4; ++it) {
                int idx = tid + it * 256;
                As[idx >> 5][idx & 31] =
                    A[(size_t)(m0 + (idx >> 5)) * lda + (k0 + (idx & 31))];
            }
#endif
        } else {
            #pragma unroll
            for (int it = 0; it < 4; ++it) {
                int idx = tid + it * 256;
                int r  = idx >> 5;
                int kk = idx & 31;
                int row = m0 + r;
                int b   = row >> 9;
                int t   = row & 511;
                int kkg = k0 + kk;
                float xv = 0.0f;
                if (kkg < 880) {
                    int kt  = kkg / 80;
                    int f   = kkg - kt * 80;
                    int tin = (t << 1) - 4 + kt;
                    if (tin >= 0 && tin < 1024)
                        xv = Xconv[((size_t)b * 1024 + tin) * 80 + f];
                }
                As[r][kk] = f2bf(xv);
            }
        }
        // ---- stage B transposed: Bt[n][k] ----
        #pragma unroll
        for (int it = 0; it < 8; ++it) {
            int idx = tid + it * 256;
            int r = idx >> 6;        // k within chunk
            int c = idx & 63;        // n within tile
            Bt[c][r] = Bm[(size_t)(k0 + r) * ldb + (n0 + c)];
        }
#if HAVE_TDM
        if (mode == 0 && wave == 0) __builtin_amdgcn_s_wait_tensorcnt(0);
#endif
        __syncthreads();

        if (kc + 1 < nk)   // prefetch next B chunk (global_prefetch_b8)
            __builtin_prefetch(&Bm[(size_t)(k0 + 32) * ldb + n0 + (tid & 63)], 0, 0);

        // ---- fragments: contiguous 16B LDS vector loads ----
        const unsigned short* ar = &As[am][0];
        v16bf af = mk_frag(*(const v8u*)&ar[akb], *(const v8u*)&ar[16 + akb]);
        const unsigned short* br = &Bt[bn_][0];
        v16bf bf = mk_frag(*(const v8u*)&br[akb], *(const v8u*)&br[16 + akb]);

        acc = __builtin_amdgcn_wmma_f32_16x16x32_bf16(
                  false, af, false, bf, (short)0, acc, false, false);
        __syncthreads();
    }

    // ---- epilogue ----
    {
        int n  = n0 + wn * 16 + (lane & 15);
        int mb = m0 + wm * 16 + ((lane >> 4) << 3);
        float g = 1.f, bt = 0.f, mn = 0.f, iv = 1.f, bs = 0.f;
        if (flags & EF_BN) {
            g = gamma[n]; bt = beta[n]; mn = mean[n]; iv = rsqrtf(var[n] + BN_EPS);
        }
        if (flags & EF_BIAS) bs = bias[n];
        #pragma unroll
        for (int i = 0; i < 8; ++i) {
            float v = acc[i] + bs;
            if (flags & EF_RELU) v = fmaxf(v, 0.0f);
            if (flags & EF_BN)   v = g * (v - mn) * iv + bt;
            size_t o = (size_t)(mb + i) * N + n;
            if (flags & EF_OUT_BF16) Cb[o] = f2bf(v);
            else                     Cf[o] = v;
        }
    }
}

// ---------------------------------------------------------------------------
// Persistent GRU scan. 32 WGs; WG i owns u-slice [16i,16i+16). Wh slice
// (512x48 bf16, stored transposed [48][512]) resident in LDS for all steps.
// Each step: h staged to LDS bf16 in 128-K chunks, rg = h @ Wh via WMMA
// (6 waves = 2 M-tiles x 3 gate-tiles), gate math, grid-wide barrier.
// ---------------------------------------------------------------------------
#define SCAN_NWG 32

__global__ __launch_bounds__(192) void gru_scan_kernel(
    const unsigned short* __restrict__ Wh,   // bf16 512 x 1536 row-major
    const float* __restrict__ brec,          // 1536
    const float* __restrict__ XG,            // 16384 x 1536 (x@Wx + b_in)
    float* __restrict__ H,                   // 32 x 512 (zero-initialized)
    float* __restrict__ Y,                   // 16384 x 512 (accumulated +=)
    unsigned int* __restrict__ bar,
    int T, int reverse)
{
    __shared__ unsigned short WhT[48][512 + 8];  // [j][k] transposed, 48.8 KB
    __shared__ unsigned short Hs[32][128 + 8];   // bf16 h chunk, 8.5 KB
    __shared__ float          Rg[32][48];        // 6 KB

    const int tid  = threadIdx.x;
    const int lane = tid & 31;
    const int wave = tid >> 5;     // 0..5
    const int wm   = wave & 1;     // M tile (batch 0-15 / 16-31)
    const int wn   = wave >> 1;    // gate tile 0..2 (z / r / h)
    const int u0   = blockIdx.x * 16;

    // Preload Wh slice transposed: local col j -> global col (j/16)*512+u0+(j%16)
    for (int idx = tid; idx < 512 * 48; idx += 192) {
        int k = idx / 48;
        int j = idx - k * 48;
        int col = (j >> 4) * 512 + u0 + (j & 15);
        WhT[j][k] = Wh[(size_t)k * 1536 + col];
    }
    __syncthreads();

    const int am  = wm * 16 + (lane & 15);
    const int akb = (lane >> 4) << 3;
    const int bn_ = wn * 16 + (lane & 15);
    const int bkb = (lane >> 4) << 4;

    unsigned int target = 0;

    for (int t = 0; t < T; ++t) {
        const int xt = reverse ? (T - 1 - t) : t;

        v8f acc = {0.f, 0.f, 0.f, 0.f, 0.f, 0.f, 0.f, 0.f};
        #pragma unroll
        for (int g = 0; g < 4; ++g) {            // 4 x 128 K halves
            if (g) __syncthreads();
            for (int idx = tid; idx < 32 * 128; idx += 192) {
                int b = idx >> 7;
                int c = idx & 127;
                Hs[b][c] = f2bf(H[(size_t)b * 512 + (g << 7) + c]);
            }
            __syncthreads();
            #pragma unroll
            for (int kc2 = 0; kc2 < 4; ++kc2) {  // 4 x 32-K WMMA chunks
                const unsigned short* ar = &Hs[am][kc2 << 5];
                v16bf af = mk_frag(*(const v8u*)&ar[akb],
                                   *(const v8u*)&ar[16 + akb]);
                const unsigned short* br = &WhT[bn_][((g << 2) + kc2) << 5];
                v16bf bf = mk_frag(*(const v8u*)&br[akb],
                                   *(const v8u*)&br[16 + akb]);
                acc = __builtin_amdgcn_wmma_f32_16x16x32_bf16(
                          false, af, false, bf, (short)0, acc, false, false);
            }
        }
        {
            int mb = wm * 16 + ((lane >> 4) << 3);
            #pragma unroll
            for (int i = 0; i < 8; ++i) Rg[mb + i][bn_] = acc[i];
        }
        __syncthreads();

        // ---- gate math for this WG's u-slice (32 batches x 16 units)
        for (int idx = tid; idx < 512; idx += 192) {
            int b = idx >> 4;
            int j = idx & 15;
            int u = u0 + j;
            size_t row = (size_t)b * T + xt;
            const float* xrow = &XG[row * 1536];
            float rz = Rg[b][j]      + brec[u];
            float rr = Rg[b][16 + j] + brec[512 + u];
            float rh = Rg[b][32 + j] + brec[1024 + u];
            float z  = 1.0f / (1.0f + __expf(-(xrow[u] + rz)));
            float r  = 1.0f / (1.0f + __expf(-(xrow[512 + u] + rr)));
            float hh = fmaxf(xrow[1024 + u] + r * rh, 0.0f);
            float hp = H[b * 512 + u];
            float hn = z * hp + (1.0f - z) * hh;
            H[b * 512 + u] = hn;
            Y[row * 512 + u] += hn;
        }

        // ---- grid-wide barrier (monotonic counter; bar zeroed per launch)
        __threadfence();
        __syncthreads();
        target += SCAN_NWG;
        if (tid == 0) {
            atomicAdd(bar, 1u);
            volatile unsigned int* vb = bar;
            while (*vb < target) { }
        }
        __syncthreads();
        __threadfence();
    }
}

// ---------------------------------------------------------------------------
// Row softmax over V columns.
// ---------------------------------------------------------------------------
__global__ __launch_bounds__(256) void softmax_kernel(
    const float* __restrict__ logits, float* __restrict__ out, int V)
{
    __shared__ float red[256];
    const size_t row = blockIdx.x;
    const float* in = logits + row * V;
    float mx = -3.0e38f;
    for (int i = threadIdx.x; i < V; i += 256) mx = fmaxf(mx, in[i]);
    red[threadIdx.x] = mx; __syncthreads();
    for (int s = 128; s > 0; s >>= 1) {
        if (threadIdx.x < s) red[threadIdx.x] = fmaxf(red[threadIdx.x], red[threadIdx.x + s]);
        __syncthreads();
    }
    mx = red[0]; __syncthreads();
    float sum = 0.f;
    for (int i = threadIdx.x; i < V; i += 256) sum += __expf(in[i] - mx);
    red[threadIdx.x] = sum; __syncthreads();
    for (int s = 128; s > 0; s >>= 1) {
        if (threadIdx.x < s) red[threadIdx.x] += red[threadIdx.x + s];
        __syncthreads();
    }
    float inv = 1.0f / red[0];
    float* o = out + row * V;
    for (int i = threadIdx.x; i < V; i += 256) o[i] = __expf(in[i] - mx) * inv;
}

// ---------------------------------------------------------------------------
// Host orchestration
// ---------------------------------------------------------------------------
extern "C" void kernel_launch(void* const* d_in, const int* in_sizes, int n_in,
                              void* d_out, int out_size, void* d_ws, size_t ws_size,
                              hipStream_t stream)
{
    (void)in_sizes; (void)n_in; (void)out_size; (void)ws_size;

    const float* x       = (const float*)d_in[0];
    const float* conv_w  = (const float*)d_in[1];
    const float* conv_b  = (const float*)d_in[2];
    const float* wx_f    = (const float*)d_in[3];
    const float* wh_f    = (const float*)d_in[4];
    const float* bi_f    = (const float*)d_in[5];
    const float* br_f    = (const float*)d_in[6];
    const float* wx_b    = (const float*)d_in[7];
    const float* wh_b    = (const float*)d_in[8];
    const float* bi_b    = (const float*)d_in[9];
    const float* br_b    = (const float*)d_in[10];
    const float* gamma   = (const float*)d_in[11];
    const float* beta    = (const float*)d_in[12];
    const float* mmean   = (const float*)d_in[13];
    const float* mvar    = (const float*)d_in[14];
    const float* dense_w = (const float*)d_in[15];
    const float* dense_b = (const float*)d_in[16];

    const int C = 512, U = 512, G = 1536, V = 1024, T2 = 512;
    const int M = 32 * T2;          // 16384 rows (b*512 + t)
    const int KCONV = 896;          // 880 padded to multiple of 32

    char* ws = (char*)d_ws;
    size_t off = 0;
    auto take = [&](size_t bytes) -> char* {
        char* p = ws + off;
        off = (off + bytes + 255) & ~(size_t)255;
        return p;
    };
    unsigned short* Wconv  = (unsigned short*)take((size_t)KCONV * C * 2);
    unsigned short* Wxf    = (unsigned short*)take((size_t)U * G * 2);
    unsigned short* Wxb    = (unsigned short*)take((size_t)U * G * 2);
    unsigned short* Whf    = (unsigned short*)take((size_t)U * G * 2);
    unsigned short* Whb    = (unsigned short*)take((size_t)U * G * 2);
    unsigned short* Wdense = (unsigned short*)take((size_t)U * V * 2);
    unsigned short* X0     = (unsigned short*)take((size_t)M * U * 2);
    float*          XG     = (float*)take((size_t)M * G * 4);
    float*          Ybuf   = (float*)take((size_t)M * U * 4);
    float*          Hbuf   = (float*)take((size_t)32 * U * 4);
    float*          Logits = (float*)take((size_t)M * V * 4);
    unsigned int*   Bar    = (unsigned int*)take(256);

    auto blocks = [](size_t n) { return (unsigned)((n + 255) / 256); };

    // 1) weights -> bf16 (conv weight zero-padded 880 -> 896 rows)
    cvt_f32_bf16_kernel<<<blocks((size_t)880 * C), 256, 0, stream>>>(conv_w, Wconv, (size_t)880 * C);
    zero_u16_kernel<<<blocks((size_t)16 * C), 256, 0, stream>>>(Wconv + (size_t)880 * C, (size_t)16 * C);
    cvt_f32_bf16_kernel<<<blocks((size_t)U * G), 256, 0, stream>>>(wx_f, Wxf, (size_t)U * G);
    cvt_f32_bf16_kernel<<<blocks((size_t)U * G), 256, 0, stream>>>(wx_b, Wxb, (size_t)U * G);
    cvt_f32_bf16_kernel<<<blocks((size_t)U * G), 256, 0, stream>>>(wh_f, Whf, (size_t)U * G);
    cvt_f32_bf16_kernel<<<blocks((size_t)U * G), 256, 0, stream>>>(wh_b, Whb, (size_t)U * G);
    cvt_f32_bf16_kernel<<<blocks((size_t)U * V), 256, 0, stream>>>(dense_w, Wdense, (size_t)U * V);

    // 2) conv (implicit im2col) + bias + relu + BN -> X0 (bf16)
    wmma_gemm_kernel<<<dim3(C / 64, M / 32), 256, 0, stream>>>(
        nullptr, x, Wconv, conv_b, nullptr, X0,
        M, C, KCONV, 0, C, /*mode=*/1,
        EF_BIAS | EF_RELU | EF_BN | EF_OUT_BF16,
        gamma, beta, mmean, mvar);

    // 3) two bidirectional GRU layers (reference reuses the same weights)
    for (int layer = 0; layer < 2; ++layer) {
        zero_f32_kernel<<<blocks((size_t)M * U), 256, 0, stream>>>(Ybuf, (size_t)M * U);
        for (int dir = 0; dir < 2; ++dir) {
            const unsigned short* Wx = dir ? Wxb : Wxf;
            const unsigned short* Wh = dir ? Whb : Whf;
            const float* bi = dir ? bi_b : bi_f;
            const float* br = dir ? br_b : br_f;
            wmma_gemm_kernel<<<dim3(G / 64, M / 32), 256, 0, stream>>>(
                X0, nullptr, Wx, bi, XG, nullptr,
                M, G, U, U, G, /*mode=*/0, EF_BIAS,
                nullptr, nullptr, nullptr, nullptr);
            zero_f32_kernel<<<blocks((size_t)32 * U), 256, 0, stream>>>(Hbuf, (size_t)32 * U);
            zero_f32_kernel<<<1, 64, 0, stream>>>((float*)Bar, 64);
            gru_scan_kernel<<<SCAN_NWG, 192, 0, stream>>>(
                Wh, br, XG, Hbuf, Ybuf, Bar, T2, dir);
        }
        if (layer == 0)
            cvt_f32_bf16_kernel<<<blocks((size_t)M * U), 256, 0, stream>>>(Ybuf, X0, (size_t)M * U);
    }

    // 4) BN -> bf16, dense GEMM, softmax
    bn_cvt_kernel<<<blocks((size_t)M * U), 256, 0, stream>>>(
        Ybuf, X0, gamma, beta, mmean, mvar, (size_t)M * U, U);
    wmma_gemm_kernel<<<dim3(V / 64, M / 32), 256, 0, stream>>>(
        X0, nullptr, Wdense, dense_b, Logits, nullptr,
        M, V, U, U, V, /*mode=*/0, EF_BIAS,
        nullptr, nullptr, nullptr, nullptr);
    softmax_kernel<<<M, 256, 0, stream>>>(Logits, (float*)d_out, V);
}
